// Attention_5927054869144
// MI455X (gfx1250) — matrix operand order repaired and checked by reading, not compiled, not run on hardware
//
#include <hip/hip_runtime.h>

typedef __bf16 bf16;
typedef __attribute__((ext_vector_type(16))) __bf16 bf16x16;
typedef __attribute__((ext_vector_type(8)))  __bf16 bf16x8;
typedef __attribute__((ext_vector_type(8)))  float  f32x8;
typedef __attribute__((ext_vector_type(4)))  int    v4i;

#define NB      8
#define CH      64
#define NHEADS  8
#define NLOW    8192
#define SPATIAL 65536
#define ATTNSCALE 0.125f

#define TCH     64      // spatial-low chunk per iteration
#define NSPLIT  16      // split-K over n (8192/16 = 512 per block)
#define XPITCH  72      // bf16 pitch of [sh][n][i] tile rows (144B, 16B-aligned, bank-skewed)
#define QKPITCH 72

// Optional CDNA5 async Global->LDS path (guarded: falls back to VGPR copy if absent)
#if defined(__has_builtin)
#if __has_builtin(__builtin_amdgcn_global_load_async_to_lds_b128) && \
    __has_builtin(__builtin_amdgcn_s_wait_asynccnt)
#define USE_ASYNC_LDS 1
#endif
#endif

// ---------------- WMMA helpers (CDNA5 gfx1250, wave32) ----------------

__device__ __forceinline__ f32x8 wmma_bf16(bf16x16 a, bf16x16 b, f32x8 c) {
  // v_wmma_f32_16x16x32_bf16 : D(16x16,f32) = A(16x32) * B(32x16) + C
  return __builtin_amdgcn_wmma_f32_16x16x32_bf16(false, a, false, b, (short)0, c, false, false);
}

// A fragment (16x32 bf16): lane m=lane%16, half=lane/16.
// Lane holds K = ks*32 + {half*8 .. half*8+7} and {16+half*8 .. +7}: two contiguous 16B runs.
__device__ __forceinline__ bf16x16 fragA(const bf16* row, int half, int ks) {
  bf16x8 lo = *(const bf16x8*)(row + ks * 32 + half * 8);
  bf16x8 hi = *(const bf16x8*)(row + ks * 32 + 16 + half * 8);
  return __builtin_shufflevector(lo, hi, 0,1,2,3,4,5,6,7,8,9,10,11,12,13,14,15);
}

// B fragment (32x16 bf16): lane col n=lane%16, half=lane/16.
// Lane holds K = ks*32 + {half*8 .. half*8+7} and {16+half*8 .. +7}, K-major storage per column.
__device__ __forceinline__ bf16x16 fragB(const bf16* col, int half, int ks) {
  bf16x8 lo = *(const bf16x8*)(col + ks * 32 + half * 8);
  bf16x8 hi = *(const bf16x8*)(col + ks * 32 + 16 + half * 8);
  return __builtin_shufflevector(lo, hi, 0,1,2,3,4,5,6,7,8,9,10,11,12,13,14,15);
}

// Stage x[b, :, sh*8192 + nbase + n] as bf16 tile sXg[sh][n][i] (i-contiguous).
// Consecutive threads take consecutive n -> 128B-coalesced global loads per i.
__device__ __forceinline__ void load_x_tile(const float* __restrict__ x, bf16* sXg,
                                            int b, int nbase, int tid) {
  for (int col = tid; col < 8 * TCH; col += 256) {
    int sh = col >> 6, n = col & 63;
    const float* src = x + (size_t)b * CH * SPATIAL + (size_t)sh * NLOW + nbase + n;
    bf16* dst = sXg + (sh * TCH + n) * XPITCH;
#pragma unroll
    for (int g = 0; g < 8; ++g) {
      bf16x8 v;
#pragma unroll
      for (int e = 0; e < 8; ++e) v[e] = (bf16)src[(size_t)(g * 8 + e) * SPATIAL];
      *(bf16x8*)(dst + g * 8) = v;
    }
  }
}

// Prefetch next chunk's x tile (global_prefetch_b8): 8 sh * 64 i rows, 256B each.
__device__ __forceinline__ void prefetch_x_tile(const float* __restrict__ x,
                                                int b, int nbase, int tid) {
  for (int r = tid; r < 8 * CH; r += 256) {
    int sh = r >> 6, i = r & 63;
    const float* p = x + (size_t)b * CH * SPATIAL + (size_t)i * SPATIAL + (size_t)sh * NLOW + nbase;
    __builtin_prefetch(p, 0, 3);
    __builtin_prefetch(p + 32, 0, 3);   // second 128B half of the 256B row
  }
}

// ---------------- Kernel 1: fused Q/K projection + dots partials ----------------
// Block = (split, head-pair hp, batch b), 256 threads = 8 waves.
// Per chunk: stage x tile -> WMMA-project Q~,K~ (A = 16 rows of Wq/Wk for heads 2hp,2hp+1)
// -> WMMA dots accumulate. Partial 2x(64x64) written per split (deterministic, no atomics).

__global__ void __launch_bounds__(256) qk_dots_kernel(
    const float* __restrict__ x,
    const float* __restrict__ Wq, const float* __restrict__ bq,
    const float* __restrict__ Wk, const float* __restrict__ bk,
    float* __restrict__ partial)   // [NSPLIT][b][head][64][64]
{
  extern __shared__ __align__(16) char smem[];
  bf16*  sW   = (bf16*)smem;                                   // [2][16][64]
  float* sb   = (float*)(smem + 4096);                         // [2][16]
  bf16*  sXg  = (bf16*)(smem + 4224);                          // [8][TCH][XPITCH]
  bf16*  sQK  = (bf16*)(smem + 4224 + 8 * TCH * XPITCH * 2);   // [2 qk][2 hl][64][QKPITCH]

  const int tid  = threadIdx.x, lane = tid & 31, wave = tid >> 5;
  const int half = lane >> 4,   l16  = lane & 15;
  const int split = blockIdx.x, hp = blockIdx.y, b = blockIdx.z;

  // Preload Wq/Wk rows hp*16 .. hp*16+15 (bf16) + biases.
  for (int t = tid; t < 2 * 16 * 64; t += 256) {
    int which = t >> 10, m = (t >> 6) & 15, i = t & 63;
    const float* W = which ? Wk : Wq;
    sW[t] = (bf16)W[(hp * 16 + m) * 64 + i];
  }
  if (tid < 32) sb[tid] = ((tid >> 4) ? bk : bq)[hp * 16 + (tid & 15)];

  const int dhl = wave >> 2, dmt = wave & 3;  // dots: wave owns (head-local, m-tile), 4 d-tiles
  f32x8 acc[4] = {};

  const int NCHUNK = (NLOW / NSPLIT) / TCH;
  for (int cc = 0; cc < NCHUNK; ++cc) {
    const int nbase = split * (NLOW / NSPLIT) + cc * TCH;
    __syncthreads();                    // protect sXg/sQK from previous iteration readers
    load_x_tile(x, sXg, b, nbase, tid);
    if (cc + 1 < NCHUNK)                // overlap next chunk's HBM latency with WMMA phase
      prefetch_x_tile(x, b, nbase + TCH, tid);
    __syncthreads();

    // Projection: 64 tiles = {q,k} x sh(8) x nsub(4); 8 tiles per wave.
#pragma unroll
    for (int t = 0; t < 8; ++t) {
      int tile = wave * 8 + t;
      int which = tile >> 5, sh = (tile >> 2) & 7, nsub = tile & 3;
      const bf16* arow = sW + which * 1024 + l16 * 64;
      bf16x16 a0 = fragA(arow, half, 0);
      bf16x16 a1 = fragA(arow, half, 1);
      const bf16* bcol = sXg + (sh * TCH + nsub * 16 + l16) * XPITCH;  // K = i (64)
      bf16x16 bb0 = fragB(bcol, half, 0);
      bf16x16 bb1 = fragB(bcol, half, 1);
      f32x8 c = {};
      c = wmma_bf16(a0, bb0, c);
      c = wmma_bf16(a1, bb1, c);
      // D element r: m = half*8 + r  ->  head-local = half, cs = r, row c = cs*8+sh
#pragma unroll
      for (int r = 0; r < 8; ++r) {
        float v = c[r] + sb[which * 16 + half * 8 + r];
        sQK[((which * 2 + half) * 64 + r * 8 + sh) * QKPITCH + nsub * 16 + l16] = (bf16)v;
      }
    }
    __syncthreads();

    // Dots: acc[dt] += Q~(rows dmt*16..)(K=n, 2 ksteps) * K~^T(cols dt*16..)
    const bf16* arow = sQK + ((0 * 2 + dhl) * 64 + dmt * 16 + l16) * QKPITCH;
    bf16x16 a0 = fragA(arow, half, 0);
    bf16x16 a1 = fragA(arow, half, 1);
#pragma unroll
    for (int dt = 0; dt < 4; ++dt) {
      const bf16* bcol = sQK + ((1 * 2 + dhl) * 64 + dt * 16 + l16) * QKPITCH;
      bf16x16 bb0 = fragB(bcol, half, 0);
      bf16x16 bb1 = fragB(bcol, half, 1);
      acc[dt] = wmma_bf16(a0, bb0, acc[dt]);
      acc[dt] = wmma_bf16(a1, bb1, acc[dt]);
    }
  }

  // Write split-partials (deterministic reduction done in kernel 2).
  const int head = hp * 2 + dhl;
  float* dst = partial + (((size_t)split * NB * NHEADS) + (size_t)b * NHEADS + head) * 4096;
#pragma unroll
  for (int dt = 0; dt < 4; ++dt)
#pragma unroll
    for (int r = 0; r < 8; ++r) {
      int ci = dmt * 16 + half * 8 + r, d = dt * 16 + l16;
      dst[ci * 64 + d] = acc[dt][r];
    }
}

// ---------------- Kernel 2: reduce + softmax + fold Wo,Wv into W2 ----------------
// One 64-thread block per (b,h). W2[b,h,sh][o][i] = sum_cs (Wo@attn)[o][cs*8+sh] * Wv[h*8+cs][i]
// const2[o] = sum_d M[o][d] * bv[h*8 + d/8]. This removes V and the output conv entirely.

__global__ void __launch_bounds__(64) softmax_fold_kernel(
    const float* __restrict__ partial, const float* __restrict__ Wo,
    const float* __restrict__ Wv, const float* __restrict__ bv,
    bf16* __restrict__ W2, float* __restrict__ c2)
{
  __shared__ float sA[64][65];
  __shared__ float sM[64][65];
  const int bh = blockIdx.x, h = bh & 7, t = threadIdx.x;

  float row[64];
  float mx = -3.4e38f;
  for (int d = 0; d < 64; ++d) {
    float s = 0.f;
    for (int sp = 0; sp < NSPLIT; ++sp)
      s += partial[((size_t)sp * 64 + bh) * 4096 + t * 64 + d];
    row[d] = s * ATTNSCALE;
    mx = fmaxf(mx, row[d]);
  }
  float s = 0.f;
  for (int d = 0; d < 64; ++d) { row[d] = expf(row[d] - mx); s += row[d]; }
  float inv = 1.f / s;
  for (int d = 0; d < 64; ++d) sA[t][d] = row[d] * inv;
  __syncthreads();

  for (int d = 0; d < 64; ++d) {       // M = Wo @ attn  (thread t = row o)
    float a = 0.f;
    for (int cc = 0; cc < 64; ++cc) a += Wo[t * 64 + cc] * sA[cc][d];
    sM[t][d] = a;
  }

  for (int sh = 0; sh < 8; ++sh)
    for (int i = 0; i < 64; ++i) {
      float a = 0.f;
#pragma unroll
      for (int cs = 0; cs < 8; ++cs) a += sM[t][cs * 8 + sh] * Wv[(h * 8 + cs) * 64 + i];
      W2[(((size_t)bh * 8 + sh) * 64 + t) * 64 + i] = (bf16)a;
    }
  float a = 0.f;
  for (int d = 0; d < 64; ++d) a += sM[t][d] * bv[h * 8 + (d >> 3)];
  c2[(size_t)bh * 64 + t] = a;
}

// ---------------- Kernel 3: out = sum_sh W2[b,h,sh] @ x_sh + const2 ----------------
// Block = (nchunk, b). x tile staged once in LDS and reused by all 8 heads (8x traffic cut);
// W2 (4MB total) streams from L2. Second full x read should largely hit the 192MB L2,
// so the 134MB of output stores are issued non-temporal to avoid evicting x.

__global__ void __launch_bounds__(256) out_proj_kernel(
    const float* __restrict__ x, const bf16* __restrict__ W2,
    const float* __restrict__ c2, float* __restrict__ out)
{
  extern __shared__ __align__(16) char smem[];
  bf16*  sXg = (bf16*)smem;                                  // [8][TCH][XPITCH]
  bf16*  sW2 = (bf16*)(smem + 8 * TCH * XPITCH * 2);         // [8][64][64]
  float* sC  = (float*)(smem + 8 * TCH * XPITCH * 2 + 8 * 64 * 64 * 2);

  const int tid  = threadIdx.x, lane = tid & 31, wave = tid >> 5;
  const int half = lane >> 4,   l16  = lane & 15;
  const int b = blockIdx.y;
  const int n0 = blockIdx.x * TCH;

  load_x_tile(x, sXg, b, n0, tid);

  const int mt  = wave & 3;
  const int nt0 = (wave >> 2) * 2, nt1 = nt0 + 1;

  for (int h = 0; h < NHEADS; ++h) {
    __syncthreads();   // x-tile ready / previous h done with sW2
    const bf16* wsrc = W2 + ((size_t)(b * NHEADS + h) * 8) * 64 * 64;   // 64KB slice
#if defined(USE_ASYNC_LDS)
    // CDNA5 async Global->LDS copy (ASYNCcnt-tracked), 16B granules.
    for (int t2 = tid; t2 < (8 * 64 * 64 * 2) / 16; t2 += 256) {
      __builtin_amdgcn_global_load_async_to_lds_b128(
          (v4i*)((const char*)wsrc + t2 * 16),
          (v4i*)((char*)sW2 + t2 * 16), 0, 0);
    }
    __builtin_amdgcn_s_wait_asynccnt(0);
#else
    {
      const unsigned* ws32 = (const unsigned*)wsrc;
      unsigned* wd32 = (unsigned*)sW2;
      for (int t2 = tid; t2 < (8 * 64 * 64) / 2; t2 += 256) wd32[t2] = ws32[t2];
    }
#endif
    if (tid < 64) sC[tid] = c2[(size_t)(b * NHEADS + h) * 64 + tid];
    __syncthreads();

    f32x8 acc0 = {}, acc1 = {};
#pragma unroll
    for (int sh = 0; sh < 8; ++sh) {
      const bf16* arow = sW2 + (sh * 64 + mt * 16 + l16) * 64;   // A = W2[sh], K = i
#pragma unroll
      for (int ks = 0; ks < 2; ++ks) {
        bf16x16 a  = fragA(arow, half, ks);
        bf16x16 f0 = fragB(sXg + (sh * TCH + nt0 * 16 + l16) * XPITCH, half, ks);
        bf16x16 f1 = fragB(sXg + (sh * TCH + nt1 * 16 + l16) * XPITCH, half, ks);
        acc0 = wmma_bf16(a, f0, acc0);
        acc1 = wmma_bf16(a, f1, acc1);
      }
    }
#pragma unroll
    for (int r = 0; r < 8; ++r) {
      int o = mt * 16 + half * 8 + r;
      float base = sC[o];
      size_t orow = ((size_t)b * CH + o) * SPATIAL + (size_t)h * NLOW + n0;
      __builtin_nontemporal_store(acc0[r] + base, &out[orow + nt0 * 16 + l16]);
      __builtin_nontemporal_store(acc1[r] + base, &out[orow + nt1 * 16 + l16]);
    }
  }
}

// ---------------- launcher ----------------

extern "C" void kernel_launch(void* const* d_in, const int* in_sizes, int n_in,
                              void* d_out, int out_size, void* d_ws, size_t ws_size,
                              hipStream_t stream) {
  (void)in_sizes; (void)n_in; (void)out_size; (void)ws_size;
  const float* x  = (const float*)d_in[0];
  const float* Wq = (const float*)d_in[1];
  const float* bq = (const float*)d_in[2];
  const float* Wk = (const float*)d_in[3];
  const float* bk = (const float*)d_in[4];
  const float* Wv = (const float*)d_in[5];
  const float* bv = (const float*)d_in[6];
  const float* Wo = (const float*)d_in[7];
  float* out = (float*)d_out;

  char* ws = (char*)d_ws;
  float* partial = (float*)ws;                                   // 16 splits * 64 * 4096 f32 = 16MB
  size_t part_bytes = (size_t)NSPLIT * NB * NHEADS * 4096 * sizeof(float);
  bf16*  W2 = (bf16*)(ws + part_bytes);                          // 64 * 8 * 4096 bf16 = 4MB
  float* c2 = (float*)(ws + part_bytes + (size_t)NB * NHEADS * 8 * 4096 * sizeof(bf16));

  const int smem1 = 4224 + 8 * TCH * XPITCH * 2 + 2 * 2 * 64 * QKPITCH * 2;     // ~114.8 KB
  const int smem3 = 8 * TCH * XPITCH * 2 + 8 * 64 * 64 * 2 + 64 * 4;            // ~136.3 KB

  qk_dots_kernel<<<dim3(NSPLIT, 4, NB), 256, smem1, stream>>>(x, Wq, bq, Wk, bk, partial);
  softmax_fold_kernel<<<dim3(NB * NHEADS), 64, 0, stream>>>(partial, Wo, Wv, bv, W2, c2);
  out_proj_kernel<<<dim3(NLOW / TCH, NB), 256, smem3, stream>>>(x, W2, c2, out);
}
